// BLNN_86199993630874
// MI455X (gfx1250) — matrix operand order repaired and checked by reading, not compiled, not run on hardware
//
#include <hip/hip_runtime.h>
#include <hip/hip_bf16.h>

#define NBLK 16
#define NTHR 128          // 4 wave32 per block
#define BB   1024
#define CC   64
#define HH   128
#define MAXIT 500
#define TOLV 0.001f

typedef _Float16 half_t;
typedef __attribute__((ext_vector_type(16))) _Float16 v16h;
typedef __attribute__((ext_vector_type(8)))  _Float16 v8h;
typedef __attribute__((ext_vector_type(8)))  float    v8f;

// ---------- fragment helpers (wave32, 16x16x32 f16 WMMA) ----------
// A 16x32 f16 layout (ISA 7.12.2): lane m=lane&15, lhi=lane>>4;
// halves 0..7 -> K = k0 + 8*lhi .. +7 ; halves 8..15 -> K = k0+16+8*lhi ..
__device__ __forceinline__ v16h ldsA(const half_t* s, int stride, int k0, int lane) {
    int m = lane & 15, lhi = lane >> 4;
    const half_t* p = s + m * stride + k0 + 8 * lhi;
    v8h lo = *(const v8h*)(p);
    v8h hi = *(const v8h*)(p + 16);
    v16h a;
#pragma unroll
    for (int j = 0; j < 8; ++j) { a[j] = lo[j]; a[j + 8] = hi[j]; }
    return a;
}

// B 32x16 f16 layout (mirrors A): lane n=lane&15, lhi=lane>>4;
// halves 0..7 -> K = k0 + 8*lhi .. +7 ; halves 8..15 -> K = k0+16+8*lhi ..
__device__ __forceinline__ v16h ldsB(const half_t* s, int stride, int k0, int n0, int lane) {
    const half_t* p = s + (k0 + 8 * (lane >> 4)) * stride + n0 + (lane & 15);
    v16h b;
#pragma unroll
    for (int j = 0; j < 8; ++j) { b[j] = p[j * stride]; b[j + 8] = p[(j + 16) * stride]; }
    return b;
}

// C/D 16x16 f32 layout: VGPR r, lane l -> (M = r + 8*(l>>4), N = l&15)
__device__ __forceinline__ void storeC16(half_t* s, int stride, int n0, v8f c, int nl, int mo) {
    int n = n0 + nl;
#pragma unroll
    for (int r = 0; r < 8; ++r) s[(r + mo) * stride + n] = (half_t)c[r];
}

__device__ __forceinline__ v8f wmma(v16h a, v16h b, v8f c) {
    return __builtin_amdgcn_wmma_f32_16x16x32_f16(false, a, false, b, (short)0, c, false, false);
}

__device__ __forceinline__ float splus(float x) {
    return x > 20.f ? x : log1pf(__expf(x));
}

// ---------- init kernel: zero the control block each launch ----------
__global__ void BLNN_init_kernel(unsigned int* ctl) {
    if (threadIdx.x < 4) ctl[threadIdx.x] = 0u;
}

// ---------- main persistent kernel ----------
__global__ void __launch_bounds__(NTHR, 1)
BLNN_86199993630874_kernel(const float* __restrict__ X,
                           const float* __restrict__ Wy0, const float* __restrict__ by0,
                           const float* __restrict__ Wy1, const float* __restrict__ by1,
                           const float* __restrict__ Wz1,
                           const float* __restrict__ Wy2, const float* __restrict__ by2,
                           const float* __restrict__ Wz2,
                           float* __restrict__ out, unsigned int* __restrict__ ctl) {
    // ---- weights in LDS (f16), both orientations for fwd/bwd GEMMs ----
    __shared__ alignas(16) half_t sWy0T[CC * HH];   // B fwd: [c][h] = Wy0[h][c]
    __shared__ alignas(16) half_t sWy0B[HH * CC];   // B bwd: [h][c] = Wy0[h][c]
    __shared__ alignas(16) half_t sWy1T[CC * HH];
    __shared__ alignas(16) half_t sWy1B[HH * CC];
    __shared__ alignas(16) half_t sWz1T[HH * HH];   // fwd: [k][n] = clip(Wz1)[n][k]
    __shared__ alignas(16) half_t sWz1B[HH * HH];   // bwd: [n][k] = clip(Wz1)[n][k] (identity)
    __shared__ float sby0[HH], sby1[HH], sWy2[CC], sWz2[HH];
    __shared__ float sby2;
    // per-wave staging (C-layout f32 -> row-major f16 A tiles)
    __shared__ alignas(16) half_t sXs[4][16 * CC];
    __shared__ alignas(16) half_t sHs[4][16 * HH];
    __shared__ alignas(16) half_t sGs[4][16 * HH];
    __shared__ int sdone;

    const int tid  = threadIdx.x;
    const int lane = tid & 31;
    const int wave = tid >> 5;
    const int nl   = lane & 15;
    const int mo   = 8 * (lane >> 4);
    const int m0   = blockIdx.x * 64 + wave * 16;   // global row base of this wave's tile

    // ---- one-time weight preload / convert / transpose ----
    for (int idx = tid; idx < HH * CC; idx += NTHR) {
        int h = idx / CC, c = idx % CC;
        float w0 = Wy0[idx], w1 = Wy1[idx];
        sWy0B[idx] = (half_t)w0; sWy0T[c * HH + h] = (half_t)w0;
        sWy1B[idx] = (half_t)w1; sWy1T[c * HH + h] = (half_t)w1;
    }
    for (int idx = tid; idx < HH * HH; idx += NTHR) {
        int n = idx / HH, k = idx % HH;
        float w = fminf(fmaxf(Wz1[idx], 0.f), 1e10f);
        sWz1B[idx] = (half_t)w;            // [n][k]
        sWz1T[k * HH + n] = (half_t)w;     // [k][n]
    }
    if (tid < HH) {
        sby0[tid] = by0[tid];
        sby1[tid] = by1[tid];
        sWz2[tid] = fminf(fmaxf(Wz2[tid], 0.f), 1e10f);
    }
    if (tid < CC) sWy2[tid] = Wy2[tid];
    if (tid == 0) { sby2 = by2[0]; sdone = 0; }
    __syncthreads();

    half_t* xs = &sXs[wave][0];
    half_t* hs = &sHs[wave][0];
    half_t* gs = &sGs[wave][0];

    unsigned int* cnt    = &ctl[0];
    unsigned int* gen    = &ctl[1];
    float*        errSum = (float*)&ctl[2];
    unsigned int* done   = &ctl[3];

    // ---- load z (input x) into C-layout registers; x1 = ones ----
    float z[4][8], x1[4][8];
#pragma unroll
    for (int t = 0; t < 4; ++t)
#pragma unroll
        for (int r = 0; r < 8; ++r) {
            z[t][r]  = X[(m0 + r + mo) * CC + 16 * t + nl];
            x1[t][r] = 1.f;
        }

    // =================== fixed-point loop ===================
    for (int it = 0; it < MAXIT; ++it) {
        // stage x1 -> f16 row-major tile
#pragma unroll
        for (int t = 0; t < 4; ++t) {
            v8f c;
#pragma unroll
            for (int r = 0; r < 8; ++r) c[r] = x1[t][r];
            storeC16(xs, CC, 16 * t, c, nl, mo);
        }

        // ---- forward layer 0: h0 = softplus(x @ Wy0^T + by0) ----
        v8f h0v[8];
#pragma unroll
        for (int t = 0; t < 8; ++t) {
            float bv = sby0[16 * t + nl];
#pragma unroll
            for (int r = 0; r < 8; ++r) h0v[t][r] = bv;
        }
#pragma unroll
        for (int kk = 0; kk < CC; kk += 32) {
            v16h a = ldsA(xs, CC, kk, lane);
#pragma unroll
            for (int t = 0; t < 8; ++t)
                h0v[t] = wmma(a, ldsB(sWy0T, HH, kk, 16 * t, lane), h0v[t]);
        }
#pragma unroll
        for (int t = 0; t < 8; ++t) {
#pragma unroll
            for (int r = 0; r < 8; ++r) h0v[t][r] = splus(h0v[t][r]);
            storeC16(hs, HH, 16 * t, h0v[t], nl, mo);
        }

        // ---- forward layer 1: h1 = softplus(h0 @ Wz1p^T + x @ Wy1^T + by1) ----
        v8f h1v[8];
#pragma unroll
        for (int t = 0; t < 8; ++t) {
            float bv = sby1[16 * t + nl];
#pragma unroll
            for (int r = 0; r < 8; ++r) h1v[t][r] = bv;
        }
#pragma unroll
        for (int kk = 0; kk < HH; kk += 32) {
            v16h a = ldsA(hs, HH, kk, lane);
#pragma unroll
            for (int t = 0; t < 8; ++t)
                h1v[t] = wmma(a, ldsB(sWz1T, HH, kk, 16 * t, lane), h1v[t]);
        }
#pragma unroll
        for (int kk = 0; kk < CC; kk += 32) {
            v16h a = ldsA(xs, CC, kk, lane);
#pragma unroll
            for (int t = 0; t < 8; ++t)
                h1v[t] = wmma(a, ldsB(sWy1T, HH, kk, 16 * t, lane), h1v[t]);
        }
#pragma unroll
        for (int t = 0; t < 8; ++t)
#pragma unroll
            for (int r = 0; r < 8; ++r) h1v[t][r] = splus(h1v[t][r]);

        // ---- output layer (scalar per row): s2 = sigmoid(h1.Wz2 + x.Wy2 + by2) ----
        float pp[8];
#pragma unroll
        for (int r = 0; r < 8; ++r) pp[r] = 0.f;
#pragma unroll
        for (int t = 0; t < 8; ++t) {
            float w = sWz2[16 * t + nl];
#pragma unroll
            for (int r = 0; r < 8; ++r) pp[r] += h1v[t][r] * w;
        }
#pragma unroll
        for (int t = 0; t < 4; ++t) {
            float w = sWy2[16 * t + nl];
#pragma unroll
            for (int r = 0; r < 8; ++r) pp[r] += x1[t][r] * w;
        }
#pragma unroll
        for (int m = 1; m < 16; m <<= 1)
#pragma unroll
            for (int r = 0; r < 8; ++r) pp[r] += __shfl_xor(pp[r], m, 32);
        float s2[8];
#pragma unroll
        for (int r = 0; r < 8; ++r) s2[r] = 1.f / (1.f + __expf(-(pp[r] + sby2)));

        // ---- backward: g_pre1 = (s2 * Wz2p) * sigmoid(pre1); sigmoid = 1-exp(-h1) ----
#pragma unroll
        for (int t = 0; t < 8; ++t) {
            float w = sWz2[16 * t + nl];
            v8f g;
#pragma unroll
            for (int r = 0; r < 8; ++r) g[r] = s2[r] * w * (1.f - __expf(-h1v[t][r]));
            storeC16(gs, HH, 16 * t, g, nl, mo);
        }

        // gx = s2*Wy2 + x1 (quadratic grad, SMOOTH=1)
        v8f gx[4];
#pragma unroll
        for (int t = 0; t < 4; ++t) {
            float w = sWy2[16 * t + nl];
#pragma unroll
            for (int r = 0; r < 8; ++r) gx[t][r] = s2[r] * w + x1[t][r];
        }

        // gx += g_pre1 @ Wy1  and  g_h0 = g_pre1 @ Wz1p
        v8f gh0[8];
#pragma unroll
        for (int t = 0; t < 8; ++t)
#pragma unroll
            for (int r = 0; r < 8; ++r) gh0[t][r] = 0.f;
#pragma unroll
        for (int kk = 0; kk < HH; kk += 32) {
            v16h a = ldsA(gs, HH, kk, lane);
#pragma unroll
            for (int t = 0; t < 4; ++t)
                gx[t] = wmma(a, ldsB(sWy1B, CC, kk, 16 * t, lane), gx[t]);
#pragma unroll
            for (int t = 0; t < 8; ++t)
                gh0[t] = wmma(a, ldsB(sWz1B, HH, kk, 16 * t, lane), gh0[t]);
        }

        // g_pre0 = g_h0 * sigmoid(pre0); stage (reuse hs)
#pragma unroll
        for (int t = 0; t < 8; ++t) {
#pragma unroll
            for (int r = 0; r < 8; ++r) gh0[t][r] *= (1.f - __expf(-h0v[t][r]));
            storeC16(hs, HH, 16 * t, gh0[t], nl, mo);
        }
        // gx += g_pre0 @ Wy0
#pragma unroll
        for (int kk = 0; kk < HH; kk += 32) {
            v16h a = ldsA(hs, HH, kk, lane);
#pragma unroll
            for (int t = 0; t < 4; ++t)
                gx[t] = wmma(a, ldsB(sWy0B, CC, kk, 16 * t, lane), gx[t]);
        }

        // ---- update + per-row error ----
        float step = 2.f / ((float)it + 1.f);
        float p[8];
#pragma unroll
        for (int r = 0; r < 8; ++r) p[r] = 0.f;
#pragma unroll
        for (int t = 0; t < 4; ++t)
#pragma unroll
            for (int r = 0; r < 8; ++r) {
                float d = z[t][r] - gx[t][r];
                x1[t][r] += step * d;
                p[r] += d * d;
            }
#pragma unroll
        for (int m = 1; m < 16; m <<= 1)
#pragma unroll
            for (int r = 0; r < 8; ++r) p[r] += __shfl_xor(p[r], m, 32);
        float te = 0.f;
#pragma unroll
        for (int r = 0; r < 8; ++r) te += sqrtf(p[r]);
        te += __shfl_xor(te, 16, 32);
        if (lane == 0) atomicAdd(errSum, te);

        // ---- grid-wide barrier + convergence decision ----
        __syncthreads();
        if (tid == 0) {
            __threadfence();
            unsigned int g = __hip_atomic_load(gen, __ATOMIC_ACQUIRE, __HIP_MEMORY_SCOPE_AGENT);
            unsigned int a = __hip_atomic_fetch_add(cnt, 1u, __ATOMIC_ACQ_REL, __HIP_MEMORY_SCOPE_AGENT);
            if (a == NBLK - 1) {
                float e = __hip_atomic_load(errSum, __ATOMIC_RELAXED, __HIP_MEMORY_SCOPE_AGENT);
                if (e < TOLV * (float)BB)
                    __hip_atomic_store(done, 1u, __ATOMIC_RELAXED, __HIP_MEMORY_SCOPE_AGENT);
                __hip_atomic_store(errSum, 0.f, __ATOMIC_RELAXED, __HIP_MEMORY_SCOPE_AGENT);
                __hip_atomic_store(cnt, 0u, __ATOMIC_RELAXED, __HIP_MEMORY_SCOPE_AGENT);
                __hip_atomic_fetch_add(gen, 1u, __ATOMIC_RELEASE, __HIP_MEMORY_SCOPE_AGENT);
            } else {
                while (__hip_atomic_load(gen, __ATOMIC_ACQUIRE, __HIP_MEMORY_SCOPE_AGENT) == g)
                    __builtin_amdgcn_s_sleep(1);
            }
            sdone = (int)__hip_atomic_load(done, __ATOMIC_RELAXED, __HIP_MEMORY_SCOPE_AGENT);
        }
        __syncthreads();
        if (sdone) break;   // done freezes x1; remaining iterations are no-ops
    }

    // ---- output: x1 + CONVEX*z ----
#pragma unroll
    for (int t = 0; t < 4; ++t)
#pragma unroll
        for (int r = 0; r < 8; ++r)
            out[(m0 + r + mo) * CC + 16 * t + nl] = x1[t][r] + z[t][r];
}

extern "C" void kernel_launch(void* const* d_in, const int* in_sizes, int n_in,
                              void* d_out, int out_size, void* d_ws, size_t ws_size,
                              hipStream_t stream) {
    (void)in_sizes; (void)n_in; (void)out_size; (void)ws_size;
    const float* X   = (const float*)d_in[0];
    const float* Wy0 = (const float*)d_in[1];
    const float* by0 = (const float*)d_in[2];
    const float* Wy1 = (const float*)d_in[3];
    const float* by1 = (const float*)d_in[4];
    const float* Wz1 = (const float*)d_in[5];
    const float* Wy2 = (const float*)d_in[6];
    const float* by2 = (const float*)d_in[7];
    const float* Wz2 = (const float*)d_in[8];
    float* out = (float*)d_out;
    unsigned int* ctl = (unsigned int*)d_ws;

    BLNN_init_kernel<<<1, 4, 0, stream>>>(ctl);
    BLNN_86199993630874_kernel<<<NBLK, NTHR, 0, stream>>>(
        X, Wy0, by0, Wy1, by1, Wz1, Wy2, by2, Wz2, out, ctl);
}
